// LSTM_GCN_2602750181730
// MI455X (gfx1250) — compile-verified, matrix operand order checked by reading, not run on hardware
//
#include <hip/hip_runtime.h>
#include <hip/hip_bf16.h>
#include <math.h>

typedef __bf16 bf16;
typedef __attribute__((ext_vector_type(16))) __bf16 v16bf;
typedef __attribute__((ext_vector_type(8)))  __bf16 v8bf;
typedef __attribute__((ext_vector_type(8)))  float  v8f;

#define B_    512
#define L_    140
#define EMB_  300
#define H_    180
#define G4_   720      // 4H (gates)
#define C2_   360      // 2H (bi concat)
#define NC_   7
#define KP1   320      // EMB padded to mult of 32
#define KP2   384      // 2H padded to mult of 32
#define KPH   192      // H padded to mult of 32
#define NT_   45       // 720/16 N-tiles
#define MROWS (B_*L_)  // 71680

union ABu { v16bf v16; v8bf v8[2]; };

// fast sigmoid/tanh built on v_exp_f32
__device__ __forceinline__ float sigf(float x)  { return 1.0f / (1.0f + __expf(-x)); }
__device__ __forceinline__ float tanhfast(float x) { return 2.0f * sigf(2.0f * x) - 1.0f; }

// K offset within a 32-wide k-block for fragment element j (0..15), lane group g (0/1)
__device__ __forceinline__ int kmap(int j, int g) {
    return (j < 8) ? (g * 8 + j) : (16 + g * 8 + (j - 8));
}

// ---------------------------------------------------------------------------
// Pack W (N=720 rows x K cols, f32 row-major) into per-lane bf16 B-fragments:
// out[((kb*45+nt)*32+lane)*16+j] = bf16(W[n][k]), n=nt*16+lane%16, k=kb*32+kmap
// ---------------------------------------------------------------------------
__global__ void pack_w_k(const float* __restrict__ W, int K, int KB, bf16* __restrict__ out) {
    int idx = blockIdx.x * 256 + threadIdx.x;
    int total = KB * NT_ * 512;
    if (idx >= total) return;
    int j    = idx & 15;
    int lane = (idx >> 4) & 31;
    int rem  = idx >> 9;
    int nt   = rem % NT_;
    int kb   = rem / NT_;
    int g    = lane >> 4;
    int k    = kb * 32 + kmap(j, g);
    int n    = nt * 16 + (lane & 15);
    float v  = (k < K) ? W[n * K + k] : 0.0f;
    out[idx] = (bf16)v;
}

// Embedding gather + cast to bf16, K padded 300->320 with zeros
__global__ void embed_k(const int* __restrict__ xadj, const float* __restrict__ emb,
                        bf16* __restrict__ xin1) {
    long long idx = (long long)blockIdx.x * 256 + threadIdx.x;
    if (idx >= (long long)MROWS * KP1) return;
    int k   = (int)(idx % KP1);
    int row = (int)(idx / KP1);
    int b = row / L_, t = row % L_;
    float v = 0.0f;
    if (k < EMB_) {
        int tok = xadj[b * (2 * L_) + t];
        v = emb[(long long)tok * EMB_ + k];
    }
    xin1[idx] = (bf16)v;
}

// Zero the 360..383 pad channels of the layer-2 input buffer
__global__ void padzero_k(bf16* __restrict__ xin2) {
    int idx = blockIdx.x * 256 + threadIdx.x;
    const int pad = KP2 - C2_;
    if (idx >= MROWS * pad) return;
    int row = idx / pad;
    int k   = C2_ + idx % pad;
    xin2[(long long)row * KP2 + k] = (bf16)0.0f;
}

// ---------------------------------------------------------------------------
// xg = X(M x Kpad, bf16) @ Wih^T + (bih + bhh)  -> (M x 720, f32)
// Each wave computes a 16x48 strip (3 N-tiles): one A-fragment load feeds
// three v_wmma_f32_16x16x32_bf16, tripling arithmetic intensity on A.
// ---------------------------------------------------------------------------
__global__ __launch_bounds__(256) void gemm_xg_k(const bf16* __restrict__ X, int Kpad,
                                                 const bf16* __restrict__ Wp,
                                                 const float* __restrict__ b0,
                                                 const float* __restrict__ b1,
                                                 float* __restrict__ xg) {
    int wave = threadIdx.x >> 5;
    int lane = threadIdx.x & 31;
    int grp  = blockIdx.x * 8 + wave;          // 8400 blocks * 8 = 67200 strips
    int mt   = grp / (NT_ / 3);                // 4480 M-tiles
    int ng   = grp % (NT_ / 3);                // 15 strip groups
    int nt0  = ng * 3;
    int l16 = lane & 15, g = lane >> 4;

    v8f acc[3];
#pragma unroll
    for (int c = 0; c < 3; ++c) {
        int ncol = (nt0 + c) * 16 + l16;
        float bsum = b0[ncol] + b1[ncol];
#pragma unroll
        for (int v = 0; v < 8; ++v) acc[c][v] = bsum;
    }
    const bf16* arow = X + (long long)(mt * 16 + l16) * Kpad;
    int KB = Kpad >> 5;
    for (int kb = 0; kb < KB; ++kb) {
        ABu a;
        const v8bf* p = (const v8bf*)(arow + kb * 32 + g * 8);
        a.v8[0] = p[0];            // K = kb*32 + g*8 .. +7
        a.v8[1] = p[2];            // K = kb*32 + 16 + g*8 .. +7
#pragma unroll
        for (int c = 0; c < 3; ++c) {
            v16bf bfrag = *(const v16bf*)(Wp + (((long long)kb * NT_ + nt0 + c) * 32 + lane) * 16);
            acc[c] = __builtin_amdgcn_wmma_f32_16x16x32_bf16(false, a.v16, false, bfrag,
                                                             (short)0, acc[c], false, false);
        }
    }
#pragma unroll
    for (int c = 0; c < 3; ++c) {
        int ncol = (nt0 + c) * 16 + l16;
#pragma unroll
        for (int v = 0; v < 8; ++v) {
            int m = mt * 16 + v + 8 * g;
            xg[(long long)m * G4_ + ncol] = acc[c][v];
        }
    }
}

// ---------------------------------------------------------------------------
// Recurrent scan: 64 blocks = 32 fwd (dir 0) + 32 bwd (dir 1), 16 batch rows
// per block, 512 threads (16 waves) to shorten the per-step critical path.
// h (bf16) and c (f32) live in LDS across all 140 steps.
// Per step: g = xg[tt] + h @ Whh^T via WMMA, then gate nonlinearity.
// ---------------------------------------------------------------------------
__global__ __launch_bounds__(512) void lstm_scan_k(const float* __restrict__ xgF,
                                                   const float* __restrict__ xgB,
                                                   const bf16* __restrict__ whhF,
                                                   const bf16* __restrict__ whhB,
                                                   float* outF, bf16* outB, int outStride) {
    __shared__ float c_s[16][H_];
    __shared__ __align__(16) bf16 hb[16][KPH];
    __shared__ float gbuf[16][G4_];
    int tid = threadIdx.x;
    int lane = tid & 31, wave = tid >> 5;
    int dir = blockIdx.x >> 5;
    int b0  = (blockIdx.x & 31) * 16;
    const float* xg  = dir ? xgB : xgF;
    const bf16*  whh = dir ? whhB : whhF;
    int chanOff = dir * H_;

    for (int p = tid; p < 16 * H_;  p += 512) c_s[p / H_][p % H_] = 0.0f;
    for (int p = tid; p < 16 * KPH; p += 512) hb[p / KPH][p % KPH] = (bf16)0.0f;
    __syncthreads();

    int l16 = lane & 15, g = lane >> 4;
    for (int t = 0; t < L_; ++t) {
        int tt = dir ? (L_ - 1 - t) : t;
        // phase A: gates = xg + h @ Whh^T   (45 N-tiles over 16 waves)
        for (int nt = wave; nt < NT_; nt += 16) {
            int ncol = nt * 16 + l16;
            v8f acc;
#pragma unroll
            for (int v = 0; v < 8; ++v) {
                int m = v + 8 * g;
                acc[v] = xg[((long long)(b0 + m) * L_ + tt) * G4_ + ncol];
            }
#pragma unroll
            for (int kb = 0; kb < KPH / 32; ++kb) {
                ABu a;
                const v8bf* p = (const v8bf*)(&hb[l16][kb * 32 + g * 8]);
                a.v8[0] = p[0];
                a.v8[1] = p[2];
                v16bf bfrag = *(const v16bf*)(whh + ((kb * NT_ + nt) * 32 + lane) * 16);
                acc = __builtin_amdgcn_wmma_f32_16x16x32_bf16(false, a.v16, false, bfrag,
                                                              (short)0, acc, false, false);
            }
#pragma unroll
            for (int v = 0; v < 8; ++v) gbuf[v + 8 * g][ncol] = acc[v];
        }
        __syncthreads();
        // phase B: LSTM cell (fast exp-based gates)
        for (int p = tid; p < 16 * H_; p += 512) {
            int m = p / H_, j = p % H_;
            float gi = gbuf[m][j];
            float gf = gbuf[m][H_ + j];
            float gg = gbuf[m][2 * H_ + j];
            float go = gbuf[m][3 * H_ + j];
            float cc = sigf(gf) * c_s[m][j] + sigf(gi) * tanhfast(gg);
            float hh = sigf(go) * tanhfast(cc);
            c_s[m][j] = cc;
            hb[m][j]  = (bf16)hh;
            long long o = ((long long)(b0 + m) * L_ + tt) * outStride + chanOff + j;
            if (outF) outF[o] = hh;
            else      outB[o] = (bf16)hh;
        }
        __syncthreads();
    }
}

// Per-t mean / rstd over (batch, channel): N = 512*360
__global__ void stats_k(const float* __restrict__ out2, float* __restrict__ meanA,
                        float* __restrict__ rstdA) {
    int t = blockIdx.x;
    __shared__ float s1[256], s2[256];
    float a = 0.f, b = 0.f;
    for (int p = threadIdx.x; p < B_ * C2_; p += 256) {
        int bb = p / C2_, ch = p % C2_;
        float v = out2[((long long)bb * L_ + t) * C2_ + ch];
        a += v; b += v * v;
    }
    s1[threadIdx.x] = a; s2[threadIdx.x] = b;
    __syncthreads();
    for (int off = 128; off > 0; off >>= 1) {
        if (threadIdx.x < off) {
            s1[threadIdx.x] += s1[threadIdx.x + off];
            s2[threadIdx.x] += s2[threadIdx.x + off];
        }
        __syncthreads();
    }
    if (threadIdx.x == 0) {
        float N = (float)(B_ * C2_);
        float mean = s1[0] / N;
        float var  = s2[0] / N - mean * mean;
        meanA[t] = mean;
        rstdA[t] = rsqrtf(var + 1e-5f);
    }
}

// normalize + relu + (360 -> 7) matmul; one wave per (b,t) row
__global__ void support_k(const float* __restrict__ out2, const float* __restrict__ meanA,
                          const float* __restrict__ rstdA, const float* __restrict__ gamma,
                          const float* __restrict__ beta, const float* __restrict__ gcw,
                          float* __restrict__ sup) {
    int wave = threadIdx.x >> 5, lane = threadIdx.x & 31;
    long long row = (long long)blockIdx.x * 8 + wave;   // b*L + t
    int t = (int)(row % L_);
    float mean = meanA[t], rstd = rstdA[t], ga = gamma[t], be = beta[t];
    float acc[NC_];
#pragma unroll
    for (int c = 0; c < NC_; ++c) acc[c] = 0.f;
    for (int j = lane; j < C2_; j += 32) {
        float v = out2[row * C2_ + j];
        v = fmaxf(ga * (v - mean) * rstd + be, 0.f);
#pragma unroll
        for (int c = 0; c < NC_; ++c) acc[c] += v * gcw[j * NC_ + c];
    }
#pragma unroll
    for (int c = 0; c < NC_; ++c) {
#pragma unroll
        for (int off = 16; off > 0; off >>= 1) acc[c] += __shfl_down(acc[c], off, 32);
    }
    if (lane == 0) {
#pragma unroll
        for (int c = 0; c < NC_; ++c) sup[row * NC_ + c] = acc[c];
    }
}

// adjacency apply + bias + median (rank 69 of 140) per (b, class); one block per b
__global__ void gcn_med_k(const float* __restrict__ sup, const int* __restrict__ xadj,
                          const float* __restrict__ gcb, float* __restrict__ out) {
    int b = blockIdx.x;
    __shared__ float s[L_][NC_];
    __shared__ float msum[NC_];
    __shared__ float garr[L_][NC_];
    int tid = threadIdx.x;
    if (tid < NC_) msum[tid] = 0.f;
    __syncthreads();
    for (int p = tid; p < L_ * NC_; p += 256) {
        int i = p / NC_, c = p % NC_;
        float v = sup[((long long)b * L_ + i) * NC_ + c];
        s[i][c] = v;
        if (xadj[b * 2 * L_ + L_ + i] == 0) atomicAdd(&msum[c], v);
    }
    __syncthreads();
    bool mlast = (xadj[b * 2 * L_ + L_ + (L_ - 1)] == 0);
    for (int p = tid; p < L_ * NC_; p += 256) {
        int i = p / NC_, c = p % NC_;
        float gv;
        if (i < L_ - 1) {
            bool mi = (xadj[b * 2 * L_ + L_ + i] == 0);
            gv = s[i][c] + (mi ? 2.f * s[L_ - 1][c] : 0.f);
        } else {
            gv = 3.f * msum[c] + (mlast ? 0.f : s[L_ - 1][c]);
        }
        garr[i][c] = gv + gcb[c];
    }
    __syncthreads();
    const int RANK = (L_ - 1) / 2;  // 69
    for (int p = tid; p < NC_ * L_; p += 256) {
        int c = p / L_, i = p % L_;
        float v = garr[i][c];
        int less = 0, eq = 0;
        for (int q = 0; q < L_; ++q) {
            float u = garr[q][c];
            less += (u < v) ? 1 : 0;
            eq   += (u == v) ? 1 : 0;
        }
        if (less <= RANK && RANK < less + eq) out[b * NC_ + c] = v;
    }
}

extern "C" void kernel_launch(void* const* d_in, const int* in_sizes, int n_in,
                              void* d_out, int out_size, void* d_ws, size_t ws_size,
                              hipStream_t stream) {
    (void)in_sizes; (void)n_in; (void)out_size; (void)ws_size;
    const int*   xadj  = (const int*)d_in[0];
    const float* emb   = (const float*)d_in[1];
    const float* Wih0f = (const float*)d_in[2];
    const float* Whh0f = (const float*)d_in[3];
    const float* bih0f = (const float*)d_in[4];
    const float* bhh0f = (const float*)d_in[5];
    const float* Wih0b = (const float*)d_in[6];
    const float* Whh0b = (const float*)d_in[7];
    const float* bih0b = (const float*)d_in[8];
    const float* bhh0b = (const float*)d_in[9];
    const float* Wih1f = (const float*)d_in[10];
    const float* Whh1f = (const float*)d_in[11];
    const float* bih1f = (const float*)d_in[12];
    const float* bhh1f = (const float*)d_in[13];
    const float* Wih1b = (const float*)d_in[14];
    const float* Whh1b = (const float*)d_in[15];
    const float* bih1b = (const float*)d_in[16];
    const float* bhh1b = (const float*)d_in[17];
    const float* gamma = (const float*)d_in[18];
    const float* beta  = (const float*)d_in[19];
    const float* gcw   = (const float*)d_in[20];
    const float* gcb   = (const float*)d_in[21];

    char* base = (char*)d_ws;
    size_t off = 0;
    auto carve = [&](size_t bytes) -> void* {
        void* p = base + off;
        off += (bytes + 255) & ~(size_t)255;
        return p;
    };
    bf16*  xin1  = (bf16*)carve((size_t)MROWS * KP1 * 2);
    bf16*  xin2  = (bf16*)carve((size_t)MROWS * KP2 * 2);
    float* xgF   = (float*)carve((size_t)MROWS * G4_ * 4);
    float* xgB   = (float*)carve((size_t)MROWS * G4_ * 4);
    float* out2  = (float*)carve((size_t)MROWS * C2_ * 4);
    float* sup   = (float*)carve((size_t)MROWS * NC_ * 4);
    float* meanA = (float*)carve(L_ * 4);
    float* rstdA = (float*)carve(L_ * 4);
    const size_t packIh1 = (size_t)(KP1 / 32) * NT_ * 512;
    const size_t packIh2 = (size_t)(KP2 / 32) * NT_ * 512;
    const size_t packHh  = (size_t)(KPH / 32) * NT_ * 512;
    bf16* pWih0f = (bf16*)carve(packIh1 * 2);
    bf16* pWih0b = (bf16*)carve(packIh1 * 2);
    bf16* pWih1f = (bf16*)carve(packIh2 * 2);
    bf16* pWih1b = (bf16*)carve(packIh2 * 2);
    bf16* pWhh0f = (bf16*)carve(packHh * 2);
    bf16* pWhh0b = (bf16*)carve(packHh * 2);
    bf16* pWhh1f = (bf16*)carve(packHh * 2);
    bf16* pWhh1b = (bf16*)carve(packHh * 2);

    // Preprocessing
    {
        long long tot = (long long)MROWS * KP1;
        embed_k<<<(int)((tot + 255) / 256), 256, 0, stream>>>(xadj, emb, xin1);
    }
    {
        int tot = MROWS * (KP2 - C2_);
        padzero_k<<<(tot + 255) / 256, 256, 0, stream>>>(xin2);
    }
    auto packLaunch = [&](const float* W, int K, int KB, bf16* outp) {
        int tot = KB * NT_ * 512;
        pack_w_k<<<(tot + 255) / 256, 256, 0, stream>>>(W, K, KB, outp);
    };
    packLaunch(Wih0f, EMB_, KP1 / 32, pWih0f);
    packLaunch(Wih0b, EMB_, KP1 / 32, pWih0b);
    packLaunch(Wih1f, C2_,  KP2 / 32, pWih1f);
    packLaunch(Wih1b, C2_,  KP2 / 32, pWih1b);
    packLaunch(Whh0f, H_,   KPH / 32, pWhh0f);
    packLaunch(Whh0b, H_,   KPH / 32, pWhh0b);
    packLaunch(Whh1f, H_,   KPH / 32, pWhh1f);
    packLaunch(Whh1b, H_,   KPH / 32, pWhh1b);

    const int gemmBlocks = (MROWS / 16) * (NT_ / 3) / 8;   // 4480*15/8 = 8400
    // Layer 1
    gemm_xg_k<<<gemmBlocks, 256, 0, stream>>>(xin1, KP1, pWih0f, bih0f, bhh0f, xgF);
    gemm_xg_k<<<gemmBlocks, 256, 0, stream>>>(xin1, KP1, pWih0b, bih0b, bhh0b, xgB);
    lstm_scan_k<<<64, 512, 0, stream>>>(xgF, xgB, pWhh0f, pWhh0b, (float*)nullptr, xin2, KP2);
    // Layer 2
    gemm_xg_k<<<gemmBlocks, 256, 0, stream>>>(xin2, KP2, pWih1f, bih1f, bhh1f, xgF);
    gemm_xg_k<<<gemmBlocks, 256, 0, stream>>>(xin2, KP2, pWih1b, bih1b, bhh1b, xgB);
    lstm_scan_k<<<64, 512, 0, stream>>>(xgF, xgB, pWhh1f, pWhh1b, out2, (bf16*)nullptr, C2_);
    // Norm + GCN head + median
    stats_k<<<L_, 256, 0, stream>>>(out2, meanA, rstdA);
    support_k<<<MROWS / 8, 256, 0, stream>>>(out2, meanA, rstdA, gamma, beta, gcw, sup);
    gcn_med_k<<<B_, 256, 0, stream>>>(sup, xadj, gcb, (float*)d_out);
}
